// MultiheadAttention_69148973466327
// MI455X (gfx1250) — matrix operand order repaired and checked by reading, not compiled, not run on hardware
//
#include <hip/hip_runtime.h>
#include <hip/hip_bf16.h>

typedef __attribute__((ext_vector_type(16))) __bf16 v16bf;
typedef __attribute__((ext_vector_type(8)))  float  v8f;
typedef int v4i __attribute__((ext_vector_type(4)));
typedef __attribute__((address_space(1))) v4i gv4i;   // global int4
typedef __attribute__((address_space(3))) v4i lv4i;   // LDS int4

__device__ __forceinline__ __bf16 f2bf(float f) { return (__bf16)f; }

__device__ __forceinline__ v8f wmma_bf16(v16bf a, v16bf b, v8f c) {
    return __builtin_amdgcn_wmma_f32_16x16x32_bf16(
        false, a, false, b, (short)0, c, false, false);
}

// pack 4 fp32 -> 4 bf16 (8 bytes) for vectorized LDS staging
__device__ __forceinline__ uint2 pack4(float a, float b, float c, float d) {
    union { __bf16 h[4]; uint2 u; } z;
    z.h[0] = (__bf16)a; z.h[1] = (__bf16)b; z.h[2] = (__bf16)c; z.h[3] = (__bf16)d;
    return z.u;
}

// A-fragment: 16-bit 16x32 layout; lane holds row lm, K chunk
// {kc+half*16 .. +16} (contiguous, matches bfrag)
__device__ __forceinline__ v16bf afrag(const __bf16* rowbase, int kc, int half) {
    v16bf a;
    const __bf16* r = rowbase + kc;
    #pragma unroll
    for (int i = 0; i < 8; ++i) a[i]     = r[half * 16 + i];
    #pragma unroll
    for (int i = 0; i < 8; ++i) a[8 + i] = r[half * 16 + 8 + i];
    return a;
}

// B-fragment when column (=N) data is K-contiguous in LDS
__device__ __forceinline__ v16bf bfrag(const __bf16* p) {
    v16bf b;
    #pragma unroll
    for (int i = 0; i < 16; ++i) b[i] = p[i];
    return b;
}

#if defined(__has_builtin)
#if __has_builtin(__builtin_amdgcn_global_load_async_to_lds_b128)
#define USE_ASYNC_LDS 1
#endif
#endif

// ---------------------------------------------------------------------------
// Kernel 1: Y = X @ W.T + bias for one 64-token x one-head (64-dim) tile,
// then (optionally) RMS-norm over D=64 + RoPE, store bf16 into [B,H,T,D].
// grid = (BT/64=128, H=16), block = 128 threads (4 waves).
// ---------------------------------------------------------------------------
__global__ __launch_bounds__(128) void proj_kernel(
    const float* __restrict__ X, const float* __restrict__ W,
    const float* __restrict__ bias, const float* __restrict__ cosp,
    const float* __restrict__ sinp, __bf16* __restrict__ dst, int do_rope)
{
    __shared__ __bf16 lA[64 * 64];     // activations tile, [row][k]
    __shared__ __bf16 lB[64 * 64];     // weight tile, [n][k] (B cols = W rows)
    __shared__ float  lY[64 * 65];     // result staging for norm/rope

    const int m0   = blockIdx.x * 64;
    const int h    = blockIdx.y;
    const int tid  = threadIdx.x;
    const int wave = tid >> 5;
    const int lane = tid & 31;
    const int half = lane >> 4;
    const int lm   = lane & 15;

    v8f acc[4] = {};

    for (int kk = 0; kk < 1024; kk += 64) {
        if (kk + 64 < 1024) {   // prefetch next K tile (global_prefetch_b8)
            __builtin_prefetch(&X[(size_t)(m0 + (tid & 63)) * 1024 + kk + 64], 0, 3);
            __builtin_prefetch(&W[(size_t)(h * 64 + (tid & 63)) * 1024 + kk + 64], 0, 3);
        }
        // stage 64x64 fp32 -> bf16: float4 loads, 8-byte packed LDS stores
        #pragma unroll
        for (int i = 0; i < 8; ++i) {
            int idx4 = i * 128 + tid;           // 0..1023 float4 units
            int r  = idx4 >> 4;                 // 16 float4 per row
            int c4 = idx4 & 15;
            const float4 xa = *(const float4*)(X + (size_t)(m0 + r) * 1024 + kk + c4 * 4);
            const float4 wb = *(const float4*)(W + (size_t)(h * 64 + r) * 1024 + kk + c4 * 4);
            *(uint2*)&lA[r * 64 + c4 * 4] = pack4(xa.x, xa.y, xa.z, xa.w);
            *(uint2*)&lB[r * 64 + c4 * 4] = pack4(wb.x, wb.y, wb.z, wb.w);
        }
        __syncthreads();

        const __bf16* arow = &lA[(wave * 16 + lm) * 64];
        v16bf a0 = afrag(arow, 0, half);
        v16bf a1 = afrag(arow, 32, half);
        #pragma unroll
        for (int t = 0; t < 4; ++t) {
            const __bf16* brow = &lB[(t * 16 + lm) * 64];
            acc[t] = wmma_bf16(a0, bfrag(brow + half * 16), acc[t]);
            acc[t] = wmma_bf16(a1, bfrag(brow + 32 + half * 16), acc[t]);
        }
        __syncthreads();
    }

    // spill C tiles (row = i + half*8, col = t*16 + lm) to LDS for row ops
    #pragma unroll
    for (int t = 0; t < 4; ++t)
        #pragma unroll
        for (int i = 0; i < 8; ++i)
            lY[(wave * 16 + i + half * 8) * 65 + t * 16 + lm] = acc[t][i];
    __syncthreads();

    if (tid < 64) {
        const int r    = tid;
        const int tokf = m0 + r;
        const int b    = tokf >> 11;       // / 2048
        const int tok  = tokf & 2047;
        float y[64];
        #pragma unroll
        for (int d = 0; d < 64; ++d) y[d] = lY[r * 65 + d] + bias[h * 64 + d];

        __bf16* o = dst + (((size_t)(b * 16 + h) * 2048 + tok) * 64);
        if (do_rope) {
            float ss = 0.f;
            #pragma unroll
            for (int d = 0; d < 64; ++d) ss += y[d] * y[d];
            const float rr = rsqrtf(ss * (1.f / 64.f) + 1e-6f);
            #pragma unroll
            for (int d = 0; d < 32; ++d) {
                float c  = cosp[tok * 32 + d];
                float s  = sinp[tok * 32 + d];
                float x1 = y[d] * rr, x2 = y[d + 32] * rr;
                o[d]      = f2bf(x1 * c - x2 * s);
                o[d + 32] = f2bf(x2 * c + x1 * s);
            }
        } else {
            #pragma unroll
            for (int d = 0; d < 64; ++d) o[d] = f2bf(y[d]);
        }
    }
}

// ---------------------------------------------------------------------------
// Kernel 2: causal flash attention for one (b,h), 64-query tile per block.
// grid = (T/64=32, B*H=64), block = 128 (4 waves; wave w owns 16 queries).
// ---------------------------------------------------------------------------
__global__ __launch_bounds__(128) void attn_kernel(
    const __bf16* __restrict__ q_ws, const __bf16* __restrict__ k_ws,
    const __bf16* __restrict__ v_ws, __bf16* __restrict__ o_ws)
{
    __shared__ __bf16 lK[64 * 64];         // [key][d]
    __shared__ __bf16 lV[64 * 64];         // [key][d]
    __shared__ __bf16 lP[4][16 * 64];      // per-wave P staging (C -> A layout)

    const int qtile = blockIdx.x;
    const int bh    = blockIdx.y;
    const int b     = bh >> 4, h = bh & 15;
    const int tid   = threadIdx.x;
    const int wave  = tid >> 5;
    const int lane  = tid & 31;
    const int half  = lane >> 4;
    const int lm    = lane & 15;
    const int qbase = qtile * 64;

    const __bf16* qp = q_ws + (size_t)bh * 2048 * 64;
    const __bf16* kp = k_ws + (size_t)bh * 2048 * 64;
    const __bf16* vp = v_ws + (size_t)bh * 2048 * 64;

    // Q fragments (row-major in memory -> direct A-fragment gathers), kept live
    v16bf qf[2];
    {
        const __bf16* qrow = qp + (size_t)(qbase + wave * 16 + lm) * 64;
        qf[0] = afrag(qrow, 0, half);
        qf[1] = afrag(qrow, 32, half);
    }

    float mrow[8], lrow[8];
    #pragma unroll
    for (int i = 0; i < 8; ++i) { mrow[i] = -3.0e38f; lrow[i] = 0.f; }
    v8f oacc[4] = {};

    const float scale = 0.125f;            // 1/sqrt(64)

    for (int j = 0; j <= qtile; ++j) {
        const __bf16* gk = kp + (size_t)j * 64 * 64;
        const __bf16* gv = vp + (size_t)j * 64 * 64;
#ifdef USE_ASYNC_LDS
        {
            const int off = tid * 8;        // bf16 elems; 16B per lane per issue
            #pragma unroll
            for (int i = 0; i < 4; ++i) {
                __builtin_amdgcn_global_load_async_to_lds_b128(
                    (gv4i*)(gk + i * 1024 + off),
                    (lv4i*)(&lK[i * 1024 + off]),
                    0, 0);
                __builtin_amdgcn_global_load_async_to_lds_b128(
                    (gv4i*)(gv + i * 1024 + off),
                    (lv4i*)(&lV[i * 1024 + off]),
                    0, 0);
            }
            asm volatile("s_wait_asynccnt 0x0" ::: "memory");
        }
#else
        #pragma unroll
        for (int i = 0; i < 4; ++i) {
            int idx8 = i * 128 + tid;       // 0..511 uint4 units (8 bf16 each)
            *(uint4*)&lK[idx8 * 8] = *(const uint4*)(gk + idx8 * 8);
            *(uint4*)&lV[idx8 * 8] = *(const uint4*)(gv + idx8 * 8);
        }
#endif
        __syncthreads();

        // S = Q @ K^T (K rows are K-contiguous columns of K^T)
        v8f sacc[4] = {};
        #pragma unroll
        for (int t = 0; t < 4; ++t) {
            const __bf16* kr = &lK[(t * 16 + lm) * 64];
            sacc[t] = wmma_bf16(qf[0], bfrag(kr + half * 16), sacc[t]);
            sacc[t] = wmma_bf16(qf[1], bfrag(kr + 32 + half * 16), sacc[t]);
        }

        const bool last = (j == qtile);
        float mnew[8];
        #pragma unroll
        for (int i = 0; i < 8; ++i) {
            float mx = -3.0e38f;
            #pragma unroll
            for (int t = 0; t < 4; ++t) {
                float v = sacc[t][i] * scale;
                if (last) {
                    int n = j * 64 + t * 16 + lm;
                    int m = qbase + wave * 16 + i + half * 8;
                    if (n > m) v = -3.0e38f;
                }
                sacc[t][i] = v;
                mx = fmaxf(mx, v);
            }
            #pragma unroll
            for (int msk = 1; msk < 16; msk <<= 1)
                mx = fmaxf(mx, __shfl_xor(mx, msk, 32));
            mnew[i] = fmaxf(mrow[i], mx);
        }

        #pragma unroll
        for (int i = 0; i < 8; ++i) {
            const float alpha = __expf(mrow[i] - mnew[i]);
            float rs = 0.f;
            #pragma unroll
            for (int t = 0; t < 4; ++t) {
                float p = __expf(sacc[t][i] - mnew[i]);
                sacc[t][i] = p;
                rs += p;
            }
            #pragma unroll
            for (int msk = 1; msk < 16; msk <<= 1)
                rs += __shfl_xor(rs, msk, 32);
            lrow[i] = lrow[i] * alpha + rs;
            mrow[i] = mnew[i];
            #pragma unroll
            for (int t = 0; t < 4; ++t) oacc[t][i] *= alpha;
        }

        // C-layout P -> A-layout via per-wave LDS (intra-wave hazard only)
        #pragma unroll
        for (int t = 0; t < 4; ++t)
            #pragma unroll
            for (int i = 0; i < 8; ++i)
                lP[wave][(i + half * 8) * 64 + t * 16 + lm] = f2bf(sacc[t][i]);
        asm volatile("s_wait_dscnt 0x0" ::: "memory");

        v16bf pf0, pf1;
        {
            const __bf16* pr = &lP[wave][lm * 64];
            pf0 = afrag(pr, 0, half);
            pf1 = afrag(pr, 32, half);
        }
        #pragma unroll
        for (int t = 0; t < 4; ++t) {
            v16bf v0, v1;
            #pragma unroll
            for (int i = 0; i < 16; ++i)
                v0[i] = lV[(half * 16 + i) * 64 + t * 16 + lm];
            #pragma unroll
            for (int i = 0; i < 16; ++i)
                v1[i] = lV[(32 + half * 16 + i) * 64 + t * 16 + lm];
            oacc[t] = wmma_bf16(pf0, v0, oacc[t]);
            oacc[t] = wmma_bf16(pf1, v1, oacc[t]);
        }
        __syncthreads();
    }

    // normalize and write [B, T, H*D] bf16 (ready for output projection)
    #pragma unroll
    for (int i = 0; i < 8; ++i) {
        const float inv = 1.0f / lrow[i];
        const int tok = qbase + wave * 16 + i + half * 8;
        #pragma unroll
        for (int t = 0; t < 4; ++t) {
            const int e = h * 64 + t * 16 + lm;
            o_ws[((size_t)(b * 2048 + tok)) * 1024 + e] = f2bf(oacc[t][i] * inv);
        }
    }
}

// ---------------------------------------------------------------------------
// Kernel 3: out = A(bf16) @ Wo.T + bo, fp32 out.
// grid = (128, 16), block = 128.
// ---------------------------------------------------------------------------
__global__ __launch_bounds__(128) void oproj_kernel(
    const __bf16* __restrict__ A, const float* __restrict__ W,
    const float* __restrict__ bias, float* __restrict__ out)
{
    __shared__ __bf16 lA[64 * 64];
    __shared__ __bf16 lB[64 * 64];

    const int m0   = blockIdx.x * 64;
    const int n0   = blockIdx.y * 64;
    const int tid  = threadIdx.x;
    const int wave = tid >> 5;
    const int lane = tid & 31;
    const int half = lane >> 4;
    const int lm   = lane & 15;

    v8f acc[4] = {};

    for (int kk = 0; kk < 1024; kk += 64) {
        if (kk + 64 < 1024)
            __builtin_prefetch(&W[(size_t)(n0 + (tid & 63)) * 1024 + kk + 64], 0, 3);
        // A is already bf16: straight 16B copies
        #pragma unroll
        for (int i = 0; i < 4; ++i) {
            int idx8 = i * 128 + tid;       // 0..511 uint4 units
            int r  = idx8 >> 3;
            int c8 = idx8 & 7;
            *(uint4*)&lA[r * 64 + c8 * 8] =
                *(const uint4*)(A + (size_t)(m0 + r) * 1024 + kk + c8 * 8);
        }
        // W: fp32 -> bf16 packed staging
        #pragma unroll
        for (int i = 0; i < 8; ++i) {
            int idx4 = i * 128 + tid;       // 0..1023 float4 units
            int r  = idx4 >> 4;
            int c4 = idx4 & 15;
            const float4 wb = *(const float4*)(W + (size_t)(n0 + r) * 1024 + kk + c4 * 4);
            *(uint2*)&lB[r * 64 + c4 * 4] = pack4(wb.x, wb.y, wb.z, wb.w);
        }
        __syncthreads();

        const __bf16* arow = &lA[(wave * 16 + lm) * 64];
        v16bf a0 = afrag(arow, 0, half);
        v16bf a1 = afrag(arow, 32, half);
        #pragma unroll
        for (int t = 0; t < 4; ++t) {
            const __bf16* brow = &lB[(t * 16 + lm) * 64];
            acc[t] = wmma_bf16(a0, bfrag(brow + half * 16), acc[t]);
            acc[t] = wmma_bf16(a1, bfrag(brow + 32 + half * 16), acc[t]);
        }
        __syncthreads();
    }

    #pragma unroll
    for (int t = 0; t < 4; ++t)
        #pragma unroll
        for (int i = 0; i < 8; ++i) {
            const int row = m0 + wave * 16 + i + half * 8;
            const int col = n0 + t * 16 + lm;
            out[(size_t)row * 1024 + col] = acc[t][i] + bias[col];
        }
}

// ---------------------------------------------------------------------------
extern "C" void kernel_launch(void* const* d_in, const int* in_sizes, int n_in,
                              void* d_out, int out_size, void* d_ws, size_t ws_size,
                              hipStream_t stream) {
    const float* q_in = (const float*)d_in[0];
    const float* k_in = (const float*)d_in[1];
    const float* v_in = (const float*)d_in[2];
    // d_in[3] = attn_mask (tril) -> causal mask implemented analytically
    const float* cosp = (const float*)d_in[4];
    const float* sinp = (const float*)d_in[5];
    const float* Wq = (const float*)d_in[6];
    const float* bq = (const float*)d_in[7];
    const float* Wk = (const float*)d_in[8];
    const float* bk = (const float*)d_in[9];
    const float* Wv = (const float*)d_in[10];
    const float* bv = (const float*)d_in[11];
    const float* Wo = (const float*)d_in[12];
    const float* bo = (const float*)d_in[13];

    const size_t NQKV = (size_t)4 * 16 * 2048 * 64;   // 8Mi elems each
    __bf16* q_ws = (__bf16*)d_ws;
    __bf16* k_ws = q_ws + NQKV;
    __bf16* v_ws = k_ws + NQKV;
    __bf16* o_ws = v_ws + NQKV;

    dim3 blk(128);
    dim3 g1(128, 16);
    proj_kernel<<<g1, blk, 0, stream>>>(q_in, Wq, bq, cosp, sinp, q_ws, 1);
    proj_kernel<<<g1, blk, 0, stream>>>(k_in, Wk, bk, cosp, sinp, k_ws, 1);
    proj_kernel<<<g1, blk, 0, stream>>>(v_in, Wv, bv, cosp, sinp, v_ws, 0);

    attn_kernel<<<dim3(32, 64), blk, 0, stream>>>(q_ws, k_ws, v_ws, o_ws);

    oproj_kernel<<<dim3(128, 16), blk, 0, stream>>>(o_ws, Wo, bo, (float*)d_out);
}
